// EMEncoder_8735963480445
// MI455X (gfx1250) — compile-verified, matrix operand order checked by reading, not run on hardware
//
#include <hip/hip_runtime.h>
#include <hip/hip_bf16.h>

// ---------------------------------------------------------------------------
// Shapes (fixed by reference setup): B=4, P=32, T=128, C=16, TC=64,
// H=256, NH=8 (head dim 32), FF=1024, V=32000.
// gat_iter=2 and n_heads=8 are Python scalars in setup_inputs(); the launch
// structure is fixed accordingly (deterministic). padding_idx is read on
// device from its input pointer.
// ---------------------------------------------------------------------------

#define Bdim 4
#define Pdim 32
#define Tdim 128
#define Cdim 16
#define TCdim 64
#define Hdim 256
#define NHdim 8
#define FFdim 1024

typedef __attribute__((ext_vector_type(16))) _Float16 v16h;
typedef __attribute__((ext_vector_type(8)))  _Float16 v8h;
typedef __attribute__((ext_vector_type(8)))  float    v8f;

// ---------------------------------------------------------------------------
// Fragment loader: per CDNA5 ISA 7.12.2, a 16-bit A/B fragment for lane L is
// two contiguous 8-half runs at K = kg and K = kg+16 (kg = (L>=16)*8).
// Loading them as two b128s directly fills the v16h in element order.
// ---------------------------------------------------------------------------
static __device__ __forceinline__ v16h load_frag16(const _Float16* __restrict__ p) {
  union { v16h v; v8h h[2]; } u;
  u.h[0] = *(const v8h*)(p);        // K = kg .. kg+7
  u.h[1] = *(const v8h*)(p + 16);   // K = kg+16 .. kg+23
  return u.v;
}

// ---------------------------------------------------------------------------
// f32 -> f16 convert with transpose: s[K,N] (row-major f32) -> d[N,K] f16.
// Used so the GEMM B operand is K-contiguous per lane (b128 fragment loads).
// ---------------------------------------------------------------------------
__global__ void cvt_transpose_kernel(const float* __restrict__ s,
                                     _Float16* __restrict__ d,
                                     int K, int N) {
  int idx = blockIdx.x * blockDim.x + threadIdx.x;
  if (idx < K * N) {
    int k = idx / N, n = idx % N;
    d[(size_t)n * K + k] = (_Float16)s[idx];
  }
}

__global__ void copy_f32_kernel(const float* __restrict__ s,
                                float* __restrict__ d, int n) {
  int i = blockIdx.x * blockDim.x + threadIdx.x;
  if (i < n) d[i] = s[i];
}

// ---------------------------------------------------------------------------
// Encode: masked embedding gather + mean pool.
// grid.x = number of nodes (B*P or B*C); block = 256 (one thread per channel)
// ---------------------------------------------------------------------------
__global__ void encode_kernel(const int* __restrict__ tokens,
                              const float* __restrict__ emb,
                              const int* __restrict__ padPtr, int NT,
                              float* __restrict__ x_out,     // [nodes,NT,H] or null
                              float* __restrict__ feat,      // [nodes,H]
                              _Float16* __restrict__ featH)  // [nodes,H]
{
  const int node = blockIdx.x;
  const int h = threadIdx.x;           // 0..255
  const int pad = padPtr[0];
  float sum = 0.f;
  int cnt = 0;
  for (int tt = 0; tt < NT; ++tt) {
    int tok = tokens[(size_t)node * NT + tt];
    bool m = (tok != pad);
    float e = m ? emb[(size_t)tok * Hdim + h] : 0.f;
    if (x_out) x_out[((size_t)node * NT + tt) * Hdim + h] = e;
    sum += e;
    cnt += m ? 1 : 0;
  }
  float f = sum / fmaxf((float)cnt, 1.f);
  feat[(size_t)node * Hdim + h] = f;
  featH[(size_t)node * Hdim + h] = (_Float16)f;
}

// ---------------------------------------------------------------------------
// WMMA f16 GEMM: out[M,N] = A[M,K] @ B[K,N], with B supplied PRE-TRANSPOSED
// as Bt[N,K] f16 row-major.  f32 accumulate via v_wmma_f32_16x16x32_f16.
//
// Block = 128 threads (4 waves); each wave owns a 16x64 output strip.
// grid = (N/64, M/64).  M%64==0, N%64==0, K%64==0 for all uses here.
//
// K-loop is an explicit unroll-by-2 ping-pong: fragment sets {a0,b0} and
// {a1,b1} are each written ONLY by global_load_b128 pairs (never copied),
// so loads land directly in the WMMA source registers while the other set's
// 4 WMMAs execute (no v_mov rotation).
//
// EPI = 0 : store f32                  (Q/K/V projections)
// EPI = 1 : +bias, relu, store f16     (FFN gemm1 -> activations)
// EPI = 2 : +bias, +resid, pad-mask, store f32 (FFN gemm2 -> para_context)
// ---------------------------------------------------------------------------
template <int EPI>
__global__ void gemm_wmma_kernel(const _Float16* __restrict__ A,   // [M,K]
                                 const _Float16* __restrict__ Bt,  // [N,K]
                                 const float* __restrict__ bias,
                                 const float* __restrict__ resid,
                                 const int* __restrict__ srcTok,
                                 const int* __restrict__ padPtr,
                                 float* __restrict__ outF,
                                 _Float16* __restrict__ outH,
                                 int M, int N, int K)
{
  const int lane = threadIdx.x & 31;
  const int wave = threadIdx.x >> 5;                 // 0..3
  const int row0 = blockIdx.y * 64 + wave * 16;
  const int col0 = blockIdx.x * 64;
  const int m    = lane & 15;
  const int kg   = (lane >> 4) * 8;                  // 0 or 8

  const _Float16* __restrict__ arow  = A + (size_t)(row0 + m) * K + kg;
  const _Float16* __restrict__ brow0 = Bt + (size_t)(col0 + m) * K + kg;

  v8f acc[4];
#pragma unroll
  for (int j = 0; j < 4; ++j)
#pragma unroll
    for (int r = 0; r < 8; ++r) acc[j][r] = 0.f;

  // prologue: phase-0 fragments at kk = 0
  v16h a0 = load_frag16(arow);
  v16h b0[4];
#pragma unroll
  for (int j = 0; j < 4; ++j) b0[j] = load_frag16(brow0 + (size_t)(j * 16) * K);

  for (int kk = 0; kk < K; kk += 64) {
    // phase-1 fragments at kk + 32
    v16h a1 = load_frag16(arow + kk + 32);
    v16h b1[4];
#pragma unroll
    for (int j = 0; j < 4; ++j)
      b1[j] = load_frag16(brow0 + (size_t)(j * 16) * K + kk + 32);

#pragma unroll
    for (int j = 0; j < 4; ++j)
      acc[j] = __builtin_amdgcn_wmma_f32_16x16x32_f16(
          false, a0, false, b0[j], (short)0, acc[j], false, false);

    if (kk + 64 < K) {
      // refill phase-0 fragments at kk + 64 (loads write a0/b0 directly)
      a0 = load_frag16(arow + kk + 64);
#pragma unroll
      for (int j = 0; j < 4; ++j)
        b0[j] = load_frag16(brow0 + (size_t)(j * 16) * K + kk + 64);
    }

#pragma unroll
    for (int j = 0; j < 4; ++j)
      acc[j] = __builtin_amdgcn_wmma_f32_16x16x32_f16(
          false, a1, false, b1[j], (short)0, acc[j], false, false);
  }

  const int mhi = (lane >> 4) << 3;                  // +0 or +8 row offset
#pragma unroll
  for (int j = 0; j < 4; ++j) {
#pragma unroll
    for (int r = 0; r < 8; ++r) {
      const int row = row0 + r + mhi;
      const int col = col0 + j * 16 + m;
      float v = acc[j][r];
      if (EPI == 1) {
        v += bias[col];
        v = v > 0.f ? v : 0.f;
        outH[(size_t)row * N + col] = (_Float16)v;
      } else if (EPI == 2) {
        v += bias[col] + resid[(size_t)row * N + col];
        if (srcTok[row] == padPtr[0]) v = 0.f;
        outF[(size_t)row * N + col] = v;
      } else {
        outF[(size_t)row * N + col] = v;
      }
    }
  }
}

// ---------------------------------------------------------------------------
// GAT attention core. One block per (b, q-node); 256 threads = 8 waves,
// wave h handles head h (head dim 32 == wave32).  Edge mask:
//   transposed==0 : mask[b][q][k] = edge[(b*Nq+q)*Nk + k]   (p2c, Nq=P,Nk=C)
//   transposed==1 : mask[b][q][k] = edge[(b*Nk+k)*Nq + q]   (c2p, Nq=C,Nk=P)
// Output: out = qn + (attn @ wo) * has_nb  (f32 + f16 copies)
// ---------------------------------------------------------------------------
__global__ void gat_attn_kernel(const float* __restrict__ Qp,
                                const float* __restrict__ Kp,
                                const float* __restrict__ Vp,
                                const float* __restrict__ qn,
                                const float* __restrict__ wo,
                                const int* __restrict__ edge,
                                int transposed, int Nq, int Nk,
                                float* __restrict__ outF,
                                _Float16* __restrict__ outH)
{
  const int blk = blockIdx.x;
  const int b = blk / Nq, qi = blk % Nq;
  const int t = threadIdx.x;          // 0..255
  const int h = t >> 5, d = t & 31;

  __shared__ float s_sc[NHdim][32];
  __shared__ float s_o[Hdim];
  __shared__ int s_any;
  if (t == 0) s_any = 0;
  __syncthreads();

  const float qv = Qp[((size_t)(b * Nq + qi)) * Hdim + h * 32 + d];

  // scores per (head, k) with wave32 dot-product reduction
  for (int k = 0; k < Nk; ++k) {
    int e = transposed ? edge[(size_t)(b * Nk + k) * Nq + qi]
                       : edge[(size_t)(b * Nq + qi) * Nk + k];
    float p = qv * Kp[((size_t)(b * Nk + k)) * Hdim + h * 32 + d];
#pragma unroll
    for (int off = 16; off; off >>= 1) p += __shfl_xor(p, off, 32);
    if (d == 0)
      s_sc[h][k] = (e > 0) ? p * 0.17677669529663687f /*1/sqrt(32)*/ : -1e9f;
    if (t == 0 && e > 0) s_any = 1;
  }
  __syncthreads();

  // softmax over k (wave h owns row h; lane index = k)
  {
    float sv = (d < Nk) ? s_sc[h][d] : -3.4e38f;
    float mx = sv;
#pragma unroll
    for (int off = 16; off; off >>= 1) mx = fmaxf(mx, __shfl_xor(mx, off, 32));
    float ex = (d < Nk) ? expf(sv - mx) : 0.f;
    float sm = ex;
#pragma unroll
    for (int off = 16; off; off >>= 1) sm += __shfl_xor(sm, off, 32);
    if (d < Nk) s_sc[h][d] = ex / sm;
  }
  __syncthreads();

  // weighted sum of V
  float o = 0.f;
  for (int k = 0; k < Nk; ++k)
    o += s_sc[h][k] * Vp[((size_t)(b * Nk + k)) * Hdim + h * 32 + d];
  s_o[t] = o;
  __syncthreads();

  // output projection (wo, f32) + residual + has_nb gating
  float acc = 0.f;
  for (int i = 0; i < Hdim; ++i) acc += s_o[i] * wo[(size_t)i * Hdim + t];
  const size_t ridx = (size_t)(b * Nq + qi) * Hdim + t;
  const float res = qn[ridx] + (s_any ? acc : 0.f);
  outF[ridx] = res;
  outH[ridx] = (_Float16)res;
}

// ---------------------------------------------------------------------------
// LayerNorm: X = para_x + state (broadcast over T); writes X back (residual)
// and LN(X)*g+b as f16 for the WMMA gemm. One block (256 thr) per row.
// ---------------------------------------------------------------------------
__global__ void ln_kernel(float* __restrict__ X,
                          const float* __restrict__ state,
                          const float* __restrict__ g,
                          const float* __restrict__ bta,
                          _Float16* __restrict__ xln)
{
  const int row = blockIdx.x;              // (b*P+p)*T + t
  const int t = threadIdx.x;               // channel
  const int node = row / Tdim;
  __shared__ float red[Hdim];

  float x = X[(size_t)row * Hdim + t] + state[(size_t)node * Hdim + t];
  X[(size_t)row * Hdim + t] = x;

  red[t] = x;
  __syncthreads();
#pragma unroll
  for (int s = Hdim / 2; s; s >>= 1) {
    if (t < s) red[t] += red[t + s];
    __syncthreads();
  }
  const float mean = red[0] / (float)Hdim;
  __syncthreads();

  const float dx = x - mean;
  red[t] = dx * dx;
  __syncthreads();
#pragma unroll
  for (int s = Hdim / 2; s; s >>= 1) {
    if (t < s) red[t] += red[t + s];
    __syncthreads();
  }
  const float var = red[0] / (float)Hdim;

  const float y = dx * rsqrtf(var + 1e-6f) * g[t] + bta[t];
  xln[(size_t)row * Hdim + t] = (_Float16)y;
}

// ---------------------------------------------------------------------------
// Host launch
// ---------------------------------------------------------------------------
extern "C" void kernel_launch(void* const* d_in, const int* in_sizes, int n_in,
                              void* d_out, int out_size, void* d_ws, size_t ws_size,
                              hipStream_t stream) {
  (void)in_sizes; (void)n_in; (void)out_size; (void)ws_size;

  const int*   src     = (const int*)d_in[0];
  const int*   cluster = (const int*)d_in[1];
  const int*   edge    = (const int*)d_in[2];
  const float* emb     = (const float*)d_in[3];
  const float* p2c_wq  = (const float*)d_in[4];
  const float* p2c_wk  = (const float*)d_in[5];
  const float* p2c_wv  = (const float*)d_in[6];
  const float* p2c_wo  = (const float*)d_in[7];
  const float* c2p_wq  = (const float*)d_in[8];
  const float* c2p_wk  = (const float*)d_in[9];
  const float* c2p_wv  = (const float*)d_in[10];
  const float* c2p_wo  = (const float*)d_in[11];
  const float* ff_w1   = (const float*)d_in[12];
  const float* ff_b1   = (const float*)d_in[13];
  const float* ff_w2   = (const float*)d_in[14];
  const float* ff_b2   = (const float*)d_in[15];
  const float* ln_g    = (const float*)d_in[16];
  const float* ln_b    = (const float*)d_in[17];
  const int*   padPtr  = (const int*)d_in[19];

  const int NTOK = Bdim * Pdim * Tdim;       // 16384 context rows
  const int MP = Bdim * Pdim;                // 128 para nodes
  const int MC = Bdim * Cdim;                // 64 cluster nodes

  // ---- workspace carve-out (~59 MB; all chunks 256B-aligned) ----
  char* wp = (char*)d_ws;
  auto carve = [&](size_t bytes) -> char* {
    char* p = wp;
    wp += (bytes + 255) & ~(size_t)255;
    return p;
  };
  float*    para_x  = (float*)   carve((size_t)NTOK * Hdim * 4);      // X (resid)
  _Float16* xln     = (_Float16*)carve((size_t)NTOK * Hdim * 2);
  _Float16* hact    = (_Float16*)carve((size_t)NTOK * FFdim * 2);
  float*    pfF     = (float*)   carve((size_t)MP * Hdim * 4);
  _Float16* pfH     = (_Float16*)carve((size_t)MP * Hdim * 2);
  float*    cfF     = (float*)   carve((size_t)MC * Hdim * 4);
  _Float16* cfH     = (_Float16*)carve((size_t)MC * Hdim * 2);
  float*    psAF    = (float*)   carve((size_t)MP * Hdim * 4);
  _Float16* psAH    = (_Float16*)carve((size_t)MP * Hdim * 2);
  float*    psBF    = (float*)   carve((size_t)MP * Hdim * 4);
  _Float16* psBH    = (_Float16*)carve((size_t)MP * Hdim * 2);
  float*    csAF    = (float*)   carve((size_t)MC * Hdim * 4);
  _Float16* csAH    = (_Float16*)carve((size_t)MC * Hdim * 2);
  float*    csBF    = (float*)   carve((size_t)MC * Hdim * 4);
  _Float16* csBH    = (_Float16*)carve((size_t)MC * Hdim * 2);
  float*    Qb      = (float*)   carve((size_t)MP * Hdim * 4);
  float*    Kb      = (float*)   carve((size_t)MP * Hdim * 4);
  float*    Vb      = (float*)   carve((size_t)MP * Hdim * 4);
  // pre-transposed f16 weights (Bt layout [N,K])
  _Float16* t_p2c_wq = (_Float16*)carve((size_t)Hdim * Hdim * 2);
  _Float16* t_p2c_wk = (_Float16*)carve((size_t)Hdim * Hdim * 2);
  _Float16* t_p2c_wv = (_Float16*)carve((size_t)Hdim * Hdim * 2);
  _Float16* t_c2p_wq = (_Float16*)carve((size_t)Hdim * Hdim * 2);
  _Float16* t_c2p_wk = (_Float16*)carve((size_t)Hdim * Hdim * 2);
  _Float16* t_c2p_wv = (_Float16*)carve((size_t)Hdim * Hdim * 2);
  _Float16* t_w1     = (_Float16*)carve((size_t)Hdim * FFdim * 2);   // [FF,H]
  _Float16* t_w2     = (_Float16*)carve((size_t)FFdim * Hdim * 2);   // [H,FF]

  auto cvtT = [&](const float* s, _Float16* d, int K, int N) {
    cvt_transpose_kernel<<<(K * N) / 256, 256, 0, stream>>>(s, d, K, N);
  };
  cvtT(p2c_wq, t_p2c_wq, Hdim, Hdim);
  cvtT(p2c_wk, t_p2c_wk, Hdim, Hdim);
  cvtT(p2c_wv, t_p2c_wv, Hdim, Hdim);
  cvtT(c2p_wq, t_c2p_wq, Hdim, Hdim);
  cvtT(c2p_wk, t_c2p_wk, Hdim, Hdim);
  cvtT(c2p_wv, t_c2p_wv, Hdim, Hdim);
  cvtT(ff_w1, t_w1, Hdim, FFdim);
  cvtT(ff_w2, t_w2, FFdim, Hdim);

  // ---- encode (para keeps context x; cluster pools only) ----
  encode_kernel<<<MP, 256, 0, stream>>>(src, emb, padPtr, Tdim, para_x, pfF, pfH);
  encode_kernel<<<MC, 256, 0, stream>>>(cluster, emb, padPtr, TCdim, nullptr, cfF, cfH);

  // ---- GAT helper: WMMA projections + attention core ----
  auto gat = [&](const float* qnF, const _Float16* qnH,
                 const _Float16* kvH,
                 const _Float16* wqt, const _Float16* wkt,
                 const _Float16* wvt, const float* woF,
                 int Nq, int Nk, int transposed,
                 float* outF, _Float16* outH) {
    const int Mq = Bdim * Nq, Mk = Bdim * Nk;
    gemm_wmma_kernel<0><<<dim3(Hdim / 64, Mq / 64), 128, 0, stream>>>(
        qnH, wqt, nullptr, nullptr, nullptr, nullptr, Qb, nullptr, Mq, Hdim, Hdim);
    gemm_wmma_kernel<0><<<dim3(Hdim / 64, Mk / 64), 128, 0, stream>>>(
        kvH, wkt, nullptr, nullptr, nullptr, nullptr, Kb, nullptr, Mk, Hdim, Hdim);
    gemm_wmma_kernel<0><<<dim3(Hdim / 64, Mk / 64), 128, 0, stream>>>(
        kvH, wvt, nullptr, nullptr, nullptr, nullptr, Vb, nullptr, Mk, Hdim, Hdim);
    gat_attn_kernel<<<Mq, 256, 0, stream>>>(Qb, Kb, Vb, qnF, woF, edge,
                                            transposed, Nq, Nk, outF, outH);
  };

  // ps0 = gat(pf, cf, p2c);  then gat_iter=2 unrolled.
  gat(pfF, pfH, cfH, t_p2c_wq, t_p2c_wk, t_p2c_wv, p2c_wo, Pdim, Cdim, 0, psAF, psAH);
  gat(cfF, cfH, psAH, t_c2p_wq, t_c2p_wk, t_c2p_wv, c2p_wo, Cdim, Pdim, 1, csAF, csAH);
  gat(psAF, psAH, csAH, t_p2c_wq, t_p2c_wk, t_p2c_wv, p2c_wo, Pdim, Cdim, 0, psBF, psBH);
  gat(csAF, csAH, psBH, t_c2p_wq, t_c2p_wk, t_c2p_wv, c2p_wo, Cdim, Pdim, 1, csBF, csBH);
  gat(psBF, psBH, csBH, t_p2c_wq, t_p2c_wk, t_p2c_wv, p2c_wo, Pdim, Cdim, 0, psAF, psAH);
  // final: para_state = psA, cluster_state = csB

  // ---- FFN over para_context ----
  ln_kernel<<<NTOK, 256, 0, stream>>>(para_x, psAF, ln_g, ln_b, xln);

  gemm_wmma_kernel<1><<<dim3(FFdim / 64, NTOK / 64), 128, 0, stream>>>(
      xln, t_w1, ff_b1, nullptr, nullptr, nullptr, nullptr, hact,
      NTOK, FFdim, Hdim);

  float* out_para_state   = (float*)d_out;
  float* out_para_context = out_para_state + (size_t)MP * Hdim;
  float* out_cluster      = out_para_context + (size_t)NTOK * Hdim;

  gemm_wmma_kernel<2><<<dim3(Hdim / 64, NTOK / 64), 128, 0, stream>>>(
      hact, t_w2, ff_b2, para_x, src, padPtr, out_para_context, nullptr,
      NTOK, Hdim, FFdim);

  // ---- emit states ----
  copy_f32_kernel<<<(MP * Hdim) / 256, 256, 0, stream>>>(psAF, out_para_state, MP * Hdim);
  copy_f32_kernel<<<(MC * Hdim) / 256, 256, 0, stream>>>(csBF, out_cluster, MC * Hdim);
}
